// BiGRU_12154757447687
// MI455X (gfx1250) — matrix operand order repaired and checked by reading, not compiled, not run on hardware
//
#include <hip/hip_runtime.h>
#include <hip/hip_bf16.h>

typedef __attribute__((ext_vector_type(16))) __bf16 v16bf;
typedef __attribute__((ext_vector_type(8)))  __bf16 v8bf;
typedef __attribute__((ext_vector_type(8)))  float  v8f;
typedef __attribute__((ext_vector_type(4)))  unsigned int v4u;
typedef __attribute__((ext_vector_type(4)))  int    v4i;
typedef __attribute__((ext_vector_type(8)))  int    v8i;

#define B_    32
#define T_    512
#define I_    512
#define H_    512
#define H3_   1536
#define MROWS (B_ * T_)   // 16384 rows for the fused input GEMM
#define NWG   8           // workgroups per direction in the scan
#define CHUNK 64          // h-columns owned per workgroup (3*CHUNK*H_ bf16 = 192 KB LDS)

__device__ __forceinline__ float sig_(float x) { return 1.0f / (1.0f + __expf(-x)); }

// A fragment 16x32 bf16: lane<16 -> M=lane, K=k0..k0+15;
// lane>=16 -> M=lane-16, K=k0+16..k0+31
__device__ __forceinline__ v16bf load_a_frag(const __bf16* __restrict__ A, int ld,
                                             int row0, int k0, int lane) {
  const int m = lane & 15, h = lane >> 4;
  const __bf16* p = A + (size_t)(row0 + m) * ld + k0 + h * 16;
  v16bf f;
  *((v8bf*)&f)     = *(const v8bf*)(p);
  *((v8bf*)&f + 1) = *(const v8bf*)(p + 8);
  return f;
}

// B fragment 32x16 bf16: lane<16 -> N=lane, K=k0..k0+15; lane>=16 -> N=lane-16, K=k0+16..k0+31
// W stored row-major [N, ld] (i.e. B = W^T), so a lane reads 16 contiguous halves.
__device__ __forceinline__ v16bf load_b_frag(const __bf16* __restrict__ W, int ld,
                                             int n0, int k0, int lane) {
  const int n = lane & 15, h = lane >> 4;
  return *(const v16bf*)(W + (size_t)(n0 + n) * ld + k0 + h * 16);
}

__device__ __forceinline__ v16bf load_b_lds(const __bf16* sm, int n0, int k0, int lane) {
  const int n = lane & 15, h = lane >> 4;
  return *(const v16bf*)(sm + (size_t)(n0 + n) * H_ + k0 + h * 16);
}

// Tensor Data Mover: stage a 2-D tile (rows x row_elems of 2-byte elements, global row
// stride row_stride_elems) into LDS at byte offset lds_byte.  D# layout per ISA 08 §8.
// This toolchain exposes the 6-arg builtin: (g0, g1, g2, g3, g4, cpol).
__device__ __forceinline__ void tdm_load_2d(unsigned lds_byte, const void* gptr,
                                            unsigned rows, unsigned row_elems,
                                            unsigned row_stride_elems) {
  const unsigned long long ga = (unsigned long long)(size_t)gptr;
  v4u g0;
  g0[0] = 1u;                                            // count=1, user descriptor
  g0[1] = lds_byte;                                      // lds_addr
  g0[2] = (unsigned)(ga & 0xFFFFFFFFu);                  // global_addr[31:0]
  g0[3] = (unsigned)((ga >> 32) & 0x1FFFFFFu) | (2u << 30); // global_addr[56:32] | type=2
  v8i g1;
  g1[0] = (int)(1u << 16);                               // data_size=1 (2 bytes), no multicast
  g1[1] = (int)((row_elems & 0xFFFFu) << 16);            // tensor_dim0[15:0]
  g1[2] = (int)(((row_elems >> 16) & 0xFFFFu) |          // tensor_dim0[31:16]
                ((rows & 0xFFFFu) << 16));               // tensor_dim1[15:0]
  g1[3] = (int)(((rows >> 16) & 0xFFFFu) |               // tensor_dim1[31:16]
                ((row_elems & 0xFFFFu) << 16));          // tile_dim0
  g1[4] = (int)(rows & 0xFFFFu);                         // tile_dim1 (tile_dim2 = 0)
  g1[5] = (int)row_stride_elems;                         // tensor_dim0_stride[31:0]
  g1[6] = 0;                                             // stride hi / dim1_stride lo
  g1[7] = 0;
  const v4i z4 = {0, 0, 0, 0};                           // 2-D tensor: groups 2/3 disabled
  const v8i z8 = {0, 0, 0, 0, 0, 0, 0, 0};
  __builtin_amdgcn_tensor_load_to_lds(g0, g1, z4, z4, z8, 0);
}

__global__ void f32_to_bf16_k(const float* __restrict__ in, __bf16* __restrict__ out, int n) {
  int i = blockIdx.x * 256 + threadIdx.x;
  if (i < n) out[i] = (__bf16)in[i];
}

__global__ void init_state_k(float* hf, __bf16* hbf, unsigned* cnt) {
  int i = blockIdx.x * 256 + threadIdx.x;
  if (i < 2 * B_ * H_)  hf[i] = 0.0f;
  if (i < 4 * B_ * H_)  hbf[i] = (__bf16)0.0f;   // 2 dirs x 2 parities
  if (i < 2)            cnt[i] = 0u;
}

// Fused input GEMM: gx[d] = src[MROWS,KI] * w_ih[d][H3_,KI]^T + b_ih[d].  One wave per 16x16 tile.
__global__ void gemm_gx(const __bf16* __restrict__ src, const __bf16* __restrict__ wih,
                        const float* __restrict__ bih, float* __restrict__ gx, int KI) {
  const int lane = threadIdx.x & 31;
  const int gw   = blockIdx.x * 8 + (threadIdx.x >> 5);
  const int per_dir = (MROWS / 16) * (H3_ / 16);   // 98304 tiles
  const int d  = gw / per_dir;
  const int r  = gw % per_dir;
  const int mt = r / (H3_ / 16);
  const int nt = r % (H3_ / 16);
  const __bf16* Wd = wih + (size_t)d * H3_ * KI;
  v8f acc = {};
  for (int k0 = 0; k0 < KI; k0 += 32) {
    v16bf a = load_a_frag(src, KI, mt * 16, k0, lane);
    v16bf b = load_b_frag(Wd,  KI, nt * 16, k0, lane);
    acc = __builtin_amdgcn_wmma_f32_16x16x32_bf16(false, a, false, b, (short)0, acc, false, false);
  }
  float* gxd = gx + (size_t)d * MROWS * H3_;
  const int c    = nt * 16 + (lane & 15);
  const float bv = bih[d * H3_ + c];
  const int mh   = (lane >> 4) * 8;                // C layout: VGPR v -> M=v (lanes 0-15) / v+8
  for (int v = 0; v < 8; ++v) {
    const int m = mt * 16 + v + mh;
    gxd[(size_t)m * H3_ + c] = acc[v] + bv;
  }
}

// Persistent recurrent scan: grid = 2 dirs * NWG workgroups, 256 threads (8 waves).
// Each WG owns CHUNK h-columns; its 3*CHUNK rows of w_hh are DMA'd into LDS by the TDM
// and stay resident for all 512 steps.
__global__ void gru_scan(const float* __restrict__ gx, const __bf16* __restrict__ whh,
                         const float* __restrict__ bhh, float* __restrict__ hf,
                         __bf16* __restrict__ hbf, unsigned* __restrict__ cnt,
                         float* __restrict__ out) {
  extern __shared__ __bf16 smem[];                 // [3*CHUNK][H_] bf16 = 192 KB
  const int d   = blockIdx.x / NWG;
  const int wg  = blockIdx.x % NWG;
  const int cb  = wg * CHUNK;
  const int tid = threadIdx.x, lane = tid & 31, wave = tid >> 5;
  const int mt  = wave >> 2, cg = wave & 3;        // 2 M-tiles x 4 column-groups = 8 waves

  // TDM preload: three 64x512 bf16 blocks of w_hh (one per gate) -> LDS.
  const __bf16* whhd = whh + (size_t)d * H3_ * H_;
  if (wave == 0) {
    const unsigned lds_base = (unsigned)(size_t)(void*)smem;
    for (int g = 0; g < 3; ++g)
      tdm_load_2d(lds_base + (unsigned)g * CHUNK * H_ * 2,
                  whhd + (size_t)(g * H_ + cb) * H_,
                  CHUNK, H_, H_);
    __builtin_amdgcn_s_wait_tensorcnt(0);
  }
  __syncthreads();

  const float* gxd = gx + (size_t)d * MROWS * H3_;
  float* hfd = hf + d * B_ * H_;
  volatile unsigned* cd = cnt + d;

  const int n  = lane & 15, mh = (lane >> 4) * 8;
  const int j  = cb + cg * 16 + n;                 // the h-column this lane owns
  const float bh_r = bhh[d * H3_ + j];
  const float bh_z = bhh[d * H3_ + H_ + j];
  const float bh_n = bhh[d * H3_ + 2 * H_ + j];

  for (int t = 0; t < T_; ++t) {
    const __bf16* hread  = hbf + (size_t)(d * 2 + (t & 1)) * B_ * H_;        // ping-pong
    __bf16*       hwrite = hbf + (size_t)(d * 2 + ((t + 1) & 1)) * B_ * H_;
    // Prefetch next step's gx rows for this lane's column while the WMMAs run.
    const int tnext = (d == 0) ? (t + 1) : (T_ - 2 - t);
    if (tnext >= 0 && tnext < T_)
      __builtin_prefetch(gxd + (size_t)((mt * 16 + mh) * T_ + tnext) * H3_ + j, 0, 1);
    v8f ar = {}, az = {}, an = {};
    for (int k0 = 0; k0 < H_; k0 += 32) {
      v16bf a  = load_a_frag(hread, H_, mt * 16, k0, lane);
      v16bf br = load_b_lds(smem, 0 * CHUNK + cg * 16, k0, lane);
      v16bf bz = load_b_lds(smem, 1 * CHUNK + cg * 16, k0, lane);
      v16bf bn = load_b_lds(smem, 2 * CHUNK + cg * 16, k0, lane);
      ar = __builtin_amdgcn_wmma_f32_16x16x32_bf16(false, a, false, br, (short)0, ar, false, false);
      az = __builtin_amdgcn_wmma_f32_16x16x32_bf16(false, a, false, bz, (short)0, az, false, false);
      an = __builtin_amdgcn_wmma_f32_16x16x32_bf16(false, a, false, bn, (short)0, an, false, false);
    }
    const int tx = (d == 0) ? t : (T_ - 1 - t);
    for (int v = 0; v < 8; ++v) {
      const int b = mt * 16 + v + mh;
      const float* gxrow = gxd + (size_t)(b * T_ + tx) * H3_;
      const float r    = sig_(gxrow[j]          + ar[v] + bh_r);
      const float z    = sig_(gxrow[H_ + j]     + az[v] + bh_z);
      const float nn   = tanhf(gxrow[2 * H_ + j] + r * (an[v] + bh_n));
      const float hold = hfd[b * H_ + j];
      const float hnew = (1.0f - z) * nn + z * hold;
      hfd[b * H_ + j]    = hnew;
      hwrite[b * H_ + j] = (__bf16)hnew;
      out[(size_t)(b * T_ + tx) * (2 * H_) + d * H_ + j] = hnew;
    }
    // per-direction grid barrier (monotone counter, NWG arrivals per step)
    __syncthreads();
    if (tid == 0) {
      __threadfence();
      atomicAdd((unsigned*)cd, 1u);
      const unsigned target = (unsigned)(NWG * (t + 1));
      while (*cd < target) { __builtin_amdgcn_s_sleep(2); }
      __threadfence();
    }
    __syncthreads();
  }
}

extern "C" void kernel_launch(void* const* d_in, const int* in_sizes, int n_in,
                              void* d_out, int out_size, void* d_ws, size_t ws_size,
                              hipStream_t stream) {
  (void)in_sizes; (void)n_in; (void)out_size; (void)ws_size;
  const float* x      = (const float*)d_in[0];
  const float* wih_l0 = (const float*)d_in[1];
  const float* whh_l0 = (const float*)d_in[2];
  const float* bih_l0 = (const float*)d_in[3];
  const float* bhh_l0 = (const float*)d_in[4];
  const float* wih_l1 = (const float*)d_in[5];
  const float* whh_l1 = (const float*)d_in[6];
  const float* bih_l1 = (const float*)d_in[7];
  const float* bhh_l1 = (const float*)d_in[8];
  float* out = (float*)d_out;

  char* ws = (char*)d_ws;
  size_t off = 0;
  auto take = [&](size_t bytes) { char* p = ws + off; off = (off + bytes + 255) & ~(size_t)255; return p; };
  __bf16*   xbf   = (__bf16*)take((size_t)MROWS * I_ * 2);          // 16 MB
  float*    h1f   = (float*) take((size_t)MROWS * 2 * H_ * 4);      // 64 MB  layer0 output
  __bf16*   h1bf  = (__bf16*)take((size_t)MROWS * 2 * H_ * 2);      // 32 MB
  __bf16*   wihbf = (__bf16*)take((size_t)2 * H3_ * (2 * H_) * 2);  // 6 MB (max KI=1024)
  __bf16*   whhbf = (__bf16*)take((size_t)2 * H3_ * H_ * 2);        // 3 MB
  float*    gxbuf = (float*) take((size_t)2 * MROWS * H3_ * 4);     // 192 MB (reused per layer)
  float*    hf    = (float*) take((size_t)2 * B_ * H_ * 4);
  __bf16*   hbf   = (__bf16*)take((size_t)4 * B_ * H_ * 2);
  unsigned* cnt   = (unsigned*)take(256);

  (void)hipFuncSetAttribute((const void*)gru_scan, hipFuncAttributeMaxDynamicSharedMemorySize,
                            3 * CHUNK * H_ * 2);

  const dim3 blk(256);
  const int gemm_blocks = 2 * (MROWS / 16) * (H3_ / 16) / 8;        // one wave per tile
  const size_t lds = (size_t)3 * CHUNK * H_ * 2;

  // ---------- layer 0 ----------
  f32_to_bf16_k<<<(MROWS * I_ + 255) / 256, blk, 0, stream>>>(x, xbf, MROWS * I_);
  f32_to_bf16_k<<<(2 * H3_ * I_ + 255) / 256, blk, 0, stream>>>(wih_l0, wihbf, 2 * H3_ * I_);
  f32_to_bf16_k<<<(2 * H3_ * H_ + 255) / 256, blk, 0, stream>>>(whh_l0, whhbf, 2 * H3_ * H_);
  gemm_gx<<<gemm_blocks, blk, 0, stream>>>(xbf, wihbf, bih_l0, gxbuf, I_);
  init_state_k<<<256, blk, 0, stream>>>(hf, hbf, cnt);
  gru_scan<<<2 * NWG, blk, lds, stream>>>(gxbuf, whhbf, bhh_l0, hf, hbf, cnt, h1f);

  // ---------- layer 1 ----------
  f32_to_bf16_k<<<(MROWS * 2 * H_ + 255) / 256, blk, 0, stream>>>(h1f, h1bf, MROWS * 2 * H_);
  f32_to_bf16_k<<<(2 * H3_ * 2 * H_ + 255) / 256, blk, 0, stream>>>(wih_l1, wihbf, 2 * H3_ * 2 * H_);
  f32_to_bf16_k<<<(2 * H3_ * H_ + 255) / 256, blk, 0, stream>>>(whh_l1, whhbf, 2 * H3_ * H_);
  gemm_gx<<<gemm_blocks, blk, 0, stream>>>(h1bf, wihbf, bih_l1, gxbuf, 2 * H_);
  init_state_k<<<256, blk, 0, stream>>>(hf, hbf, cnt);
  gru_scan<<<2 * NWG, blk, lds, stream>>>(gxbuf, whhbf, bhh_l1, hf, hbf, cnt, out);
}
